// SAGE_51032801411846
// MI455X (gfx1250) — matrix operand order repaired and checked by reading, not compiled, not run on hardware
//
#include <hip/hip_runtime.h>
#include <hip/hip_bf16.h>

// ---------------------------------------------------------------------------
// GCN (2x GCNConv + BN + Linear + log_softmax + var) for gfx1250.
// GEMMs: v_wmma_f32_16x16x32_bf16, weights pre-packed into fragment order so
// every operand load is a coalesced b128. Aggregation: wave-per-edge float4
// gather + f32 atomic scatter (hw+agg ~174MB -> L2-resident on 192MB L2).
// ---------------------------------------------------------------------------

typedef __attribute__((ext_vector_type(16))) __bf16 v16bf;
typedef __attribute__((ext_vector_type(8)))  float  v8f;

struct Frag { // 16 bf16 = 32 bytes, loaded as two 16B quads
    union { v16bf v; uint4 q[2]; unsigned short u[16]; };
};

__device__ __forceinline__ unsigned short f32_to_bf16(float f) {
    unsigned int u = __float_as_uint(f);
    u += 0x7FFFu + ((u >> 16) & 1u);        // round-to-nearest-even
    return (unsigned short)(u >> 16);
}

// ---------------------------- small elementwise ----------------------------

__global__ void k_fill_f32(float* __restrict__ p, float v, int n) {
    int i = blockIdx.x * blockDim.x + threadIdx.x;
    if (i < n) p[i] = v;
}

__global__ void k_deg_edges(const int* __restrict__ dst, float* __restrict__ deg, int e) {
    int i = blockIdx.x * blockDim.x + threadIdx.x;
    if (i < e) atomicAdd(&deg[dst[i]], 1.0f);
}

__global__ void k_dinv(float* __restrict__ deg, int n) {
    int i = blockIdx.x * blockDim.x + threadIdx.x;
    if (i < n) deg[i] = rsqrtf(fmaxf(deg[i], 1.0f));
}

__global__ void k_to_bf16(const float* __restrict__ in, unsigned short* __restrict__ out, long n) {
    long i = (long)blockIdx.x * blockDim.x + threadIdx.x;
    if (i < n) out[i] = f32_to_bf16(in[i]);
}

// Pack weight W[128, Cin] (f32, row-major) into WMMA B-fragment order:
//   packed[((nt*4 + kbi)*32 + lane)*16 + e] = W[(kbi*32 + (e>>3)*16 + (lane>>4)*8 + (e&7)), nt*16 + (lane&15)]
// (zero-pads columns >= Cin, e.g. W3 40 -> 48). One thread per packed element.
__global__ void k_pack_w_bf16(const float* __restrict__ w, unsigned short* __restrict__ out,
                              int cin, int ntiles) {
    int i = blockIdx.x * blockDim.x + threadIdx.x;
    if (i >= ntiles * 2048) return;            // ntiles * 4 kb * 32 lanes * 16 elems
    int e    = i & 15;
    int lane = (i >> 4) & 31;
    int kbi  = (i >> 9) & 3;
    int nt   = i >> 11;
    int k    = kbi * 32 + (e >> 3) * 16 + (lane >> 4) * 8 + (e & 7);
    int col  = nt * 16 + (lane & 15);
    float v  = (col < cin) ? w[k * cin + col] : 0.0f;
    out[i]   = f32_to_bf16(v);
}

// ------------------------------- WMMA GEMM ---------------------------------
// C[M, Nstore] = A[M,128](bf16 row-major) x Wp(fragment-packed, NT n-tiles) + bias.
// One wave per 16-row strip; A fragment reused across all NT output tiles.

template <int NT>
__global__ void k_gemm_bf16_wmma(const unsigned short* __restrict__ A,
                                 const unsigned short* __restrict__ Bp,
                                 const float* __restrict__ bias,
                                 float* __restrict__ Cm,
                                 int M, int Nstore) {
    const int K    = 128;
    const int wave = threadIdx.x >> 5;                 // 4 waves / block
    const int tile_m = (blockIdx.x * 4 + wave) * 16;
    if (tile_m >= M) return;                           // uniform per wave
    const int lane = threadIdx.x & 31;
    const int half = lane >> 4;                        // 0: lanes 0-15, 1: 16-31
    const int lrow = lane & 15;

    int arow = tile_m + lrow;
    if (arow >= M) arow = M - 1;                       // clamp; discarded at store
    const unsigned short* Arow = A + (long)arow * K;

    v8f acc[NT] = {};
    #pragma unroll
    for (int kbi = 0; kbi < 4; ++kbi) {
        const int kb = kbi * 32;
        // A 16x32 bf16 fragment (ISA 7.12.2): two contiguous 16B quads per lane.
        Frag a;
        a.q[0] = *(const uint4*)(Arow + kb + half * 8);
        a.q[1] = *(const uint4*)(Arow + kb + 16 + half * 8);
        #pragma unroll
        for (int nt = 0; nt < NT; ++nt) {
            // Pre-packed B fragment: 32 contiguous bytes per lane, coalesced.
            const unsigned short* bp = Bp + (((nt * 4 + kbi) * 32 + lane) << 4);
            Frag b;
            b.q[0] = *(const uint4*)(bp);
            b.q[1] = *(const uint4*)(bp + 8);
            acc[nt] = __builtin_amdgcn_wmma_f32_16x16x32_bf16(
                false, a.v, false, b.v, (short)0, acc[nt], false, false);
        }
    }
    // C/D 16x16 f32 layout: VGPR r -> row (r + half*8), lane%16 -> col.
    #pragma unroll
    for (int nt = 0; nt < NT; ++nt) {
        #pragma unroll
        for (int r = 0; r < 8; ++r) {
            int row = tile_m + half * 8 + r;
            int col = nt * 16 + lrow;
            if (row < M && col < Nstore) {
                float v = acc[nt][r];
                if (bias) v += bias[col];
                Cm[(long)row * Nstore + col] = v;
            }
        }
    }
}

// ----------------------------- GCN aggregation -----------------------------

// agg[i][j] = b[j] + hw[i][j] * dinv[i]^2   (bias + self-loop term)
__global__ void k_agg_init(const float* __restrict__ hw, const float* __restrict__ dinv,
                           const float* __restrict__ bias, float* __restrict__ agg, long total) {
    long i = (long)blockIdx.x * blockDim.x + threadIdx.x;
    if (i >= total) return;
    int node = (int)(i >> 7);
    int j    = (int)(i & 127);
    float di = dinv[node];
    agg[i] = bias[j] + hw[i] * di * di;
}

// wave-per-edge: float4 gather of hw[src], scale by dinv[src]*dinv[dst], atomic scatter
__global__ void k_agg_edges(const int* __restrict__ src, const int* __restrict__ dst,
                            const float* __restrict__ dinv,
                            const float* __restrict__ hw, float* __restrict__ agg, int e) {
    long t   = (long)blockIdx.x * blockDim.x + threadIdx.x;
    int eid  = (int)(t >> 5);
    int lane = (int)(t & 31);
    if (eid >= e) return;
    int s = src[eid], d = dst[eid];
    float w = dinv[s] * dinv[d];
    const float4 hv = *(const float4*)(hw + (long)s * 128 + lane * 4);
    float* dp = agg + (long)d * 128 + lane * 4;
    atomicAdd(dp + 0, hv.x * w);
    atomicAdd(dp + 1, hv.y * w);
    atomicAdd(dp + 2, hv.z * w);
    atomicAdd(dp + 3, hv.w * w);
}

// relu -> BN(eval) -> relu -> bf16 (next layer A matrix)
__global__ void k_relu_bn_relu_bf16(const float* __restrict__ agg,
                                    const float* __restrict__ g, const float* __restrict__ be,
                                    const float* __restrict__ mu, const float* __restrict__ var,
                                    unsigned short* __restrict__ hb, long total) {
    long i = (long)blockIdx.x * blockDim.x + threadIdx.x;
    if (i >= total) return;
    int j = (int)(i & 127);
    float t = fmaxf(agg[i], 0.0f);
    t = (t - mu[j]) * rsqrtf(var[j] + 1e-5f) * g[j] + be[j];
    hb[i] = f32_to_bf16(fmaxf(t, 0.0f));
}

// ------------------------------ epilogue -----------------------------------

__global__ void k_var_zero(double* __restrict__ s) { s[0] = 0.0; s[1] = 0.0; }

__global__ void k_var_accum(const float* __restrict__ x, double* __restrict__ s, long total) {
    __shared__ double ssum[256], ssq[256];
    long i = (long)blockIdx.x * blockDim.x + threadIdx.x;
    long stride = (long)gridDim.x * blockDim.x;
    double a = 0.0, b = 0.0;
    for (; i < total; i += stride) { double v = (double)x[i]; a += v; b += v * v; }
    ssum[threadIdx.x] = a; ssq[threadIdx.x] = b;
    __syncthreads();
    for (int o = blockDim.x >> 1; o > 0; o >>= 1) {
        if ((int)threadIdx.x < o) {
            ssum[threadIdx.x] += ssum[threadIdx.x + o];
            ssq[threadIdx.x]  += ssq[threadIdx.x + o];
        }
        __syncthreads();
    }
    if (threadIdx.x == 0) { atomicAdd(&s[0], ssum[0]); atomicAdd(&s[1], ssq[0]); }
}

__global__ void k_var_final(const double* __restrict__ s, float* __restrict__ outv, double total) {
    double mean = s[0] / total;
    *outv = (float)((s[1] - total * mean * mean) / (total - 1.0));
}

// in-place log_softmax, one wave per row of [n, c], c <= 64
__global__ void k_log_softmax(float* __restrict__ out, int n, int c) {
    long t   = (long)blockIdx.x * blockDim.x + threadIdx.x;
    int row  = (int)(t >> 5);
    int lane = (int)(t & 31);
    if (row >= n) return;
    float* p = out + (long)row * c;
    float ninf = -__builtin_inff();
    float x0 = (lane      < c) ? p[lane]      : ninf;
    float x1 = (lane + 32 < c) ? p[lane + 32] : ninf;
    float m = fmaxf(x0, x1);
    #pragma unroll
    for (int off = 16; off > 0; off >>= 1) m = fmaxf(m, __shfl_xor(m, off, 32));
    float s = 0.0f;
    if (lane      < c) s += __expf(x0 - m);
    if (lane + 32 < c) s += __expf(x1 - m);
    #pragma unroll
    for (int off = 16; off > 0; off >>= 1) s += __shfl_xor(s, off, 32);
    float lse = m + __logf(s);
    if (lane      < c) p[lane]      = x0 - lse;
    if (lane + 32 < c) p[lane + 32] = x1 - lse;
}

// ---------------------------------------------------------------------------

static inline size_t alignup(size_t v) { return (v + 255) & ~(size_t)255; }

extern "C" void kernel_launch(void* const* d_in, const int* in_sizes, int n_in,
                              void* d_out, int out_size, void* d_ws, size_t ws_size,
                              hipStream_t stream) {
    const float* x  = (const float*)d_in[0];
    const int*   ei = (const int*)d_in[1];
    const float* W1 = (const float*)d_in[2];
    const float* b1 = (const float*)d_in[3];
    const float* g1 = (const float*)d_in[4];
    const float* be1= (const float*)d_in[5];
    const float* m1 = (const float*)d_in[6];
    const float* v1 = (const float*)d_in[7];
    const float* W2 = (const float*)d_in[8];
    const float* b2 = (const float*)d_in[9];
    const float* g2 = (const float*)d_in[10];
    const float* be2= (const float*)d_in[11];
    const float* m2 = (const float*)d_in[12];
    const float* v2 = (const float*)d_in[13];
    const float* W3 = (const float*)d_in[14];
    const float* b3 = (const float*)d_in[15];

    const int K  = 128, Hd = 128, C = 40;
    const int n  = in_sizes[0] / K;   // nodes
    const int e  = in_sizes[1] / 2;   // edges
    const int* src = ei;
    const int* dst = ei + e;

    // workspace carve-up
    char* ws = (char*)d_ws;
    size_t off = 0;
    float*          dinv = (float*)(ws + off);          off = alignup(off + (size_t)n * 4);
    unsigned short* xb   = (unsigned short*)(ws + off); off = alignup(off + (size_t)n * K * 2);
    float*          hw   = (float*)(ws + off);          off = alignup(off + (size_t)n * Hd * 4);
    float*          agg  = (float*)(ws + off);          off = alignup(off + (size_t)n * Hd * 4);
    unsigned short* W1p  = (unsigned short*)(ws + off); off = alignup(off + (size_t)8 * 2048 * 2);
    unsigned short* W2p  = (unsigned short*)(ws + off); off = alignup(off + (size_t)8 * 2048 * 2);
    unsigned short* W3p  = (unsigned short*)(ws + off); off = alignup(off + (size_t)3 * 2048 * 2);
    double*         sums = (double*)(ws + off);         off = alignup(off + 2 * sizeof(double));
    (void)ws_size; (void)n_in; (void)out_size;

    float* out = (float*)d_out;
    const long totNH = (long)n * Hd;

    const int TB = 256;
    auto blks = [&](long work) { return (unsigned)((work + TB - 1) / TB); };

    // --- degree / normalization ---
    k_fill_f32<<<blks(n), TB, 0, stream>>>(dinv, 1.0f, n);           // self-loop
    k_deg_edges<<<blks(e), TB, 0, stream>>>(dst, dinv, e);
    k_dinv<<<blks(n), TB, 0, stream>>>(dinv, n);
    k_var_zero<<<1, 1, 0, stream>>>(sums);

    // --- bf16 conversion / weight fragment packing ---
    k_to_bf16<<<blks(totNH), TB, 0, stream>>>(x, xb, totNH);
    k_pack_w_bf16<<<blks(8 * 2048), TB, 0, stream>>>(W1, W1p, Hd, 8);
    k_pack_w_bf16<<<blks(8 * 2048), TB, 0, stream>>>(W2, W2p, Hd, 8);
    k_pack_w_bf16<<<blks(3 * 2048), TB, 0, stream>>>(W3, W3p, C,  3);

    dim3 gemm_blk(128, 1, 1);                     // 4 waves, one 16-row strip each
    dim3 gemm_grd((n + 63) / 64, 1, 1);

    // --- layer 1: hw = x@W1 ; aggregate ; relu/bn/relu -> xb ---
    k_gemm_bf16_wmma<8><<<gemm_grd, gemm_blk, 0, stream>>>(xb, W1p, nullptr, hw, n, Hd);
    k_agg_init<<<blks(totNH), TB, 0, stream>>>(hw, dinv, b1, agg, totNH);
    k_agg_edges<<<blks((long)e * 32), TB, 0, stream>>>(src, dst, dinv, hw, agg, e);
    k_relu_bn_relu_bf16<<<blks(totNH), TB, 0, stream>>>(agg, g1, be1, m1, v1, xb, totNH);

    // --- layer 2 ---
    k_gemm_bf16_wmma<8><<<gemm_grd, gemm_blk, 0, stream>>>(xb, W2p, nullptr, hw, n, Hd);
    k_agg_init<<<blks(totNH), TB, 0, stream>>>(hw, dinv, b2, agg, totNH);
    k_agg_edges<<<blks((long)e * 32), TB, 0, stream>>>(src, dst, dinv, hw, agg, e);
    k_relu_bn_relu_bf16<<<blks(totNH), TB, 0, stream>>>(agg, g2, be2, m2, v2, xb, totNH);

    // --- final linear straight into d_out ---
    k_gemm_bf16_wmma<3><<<gemm_grd, gemm_blk, 0, stream>>>(xb, W3p, b3, out, n, C);

    // --- variance (before in-place softmax), then log_softmax ---
    const long totNC = (long)n * C;
    k_var_accum<<<1024, TB, 0, stream>>>(out, sums, totNC);
    k_var_final<<<1, 1, 0, stream>>>(sums, out + totNC, (double)totNC);
    k_log_softmax<<<blks((long)n * 32), TB, 0, stream>>>(out, n, C);
}
